// TemporalAttention_45818711114210
// MI455X (gfx1250) — matrix operand order repaired and checked by reading, not compiled, not run on hardware
//
#include <hip/hip_runtime.h>

typedef __bf16 bf16_t;
typedef __attribute__((ext_vector_type(16))) __bf16 v16bf;
typedef __attribute__((ext_vector_type(8)))  __bf16 v8bf;
typedef __attribute__((ext_vector_type(8)))  float  v8f;
typedef __attribute__((ext_vector_type(4))) unsigned int v4u;
typedef __attribute__((ext_vector_type(8))) int v8i_t;
typedef __attribute__((ext_vector_type(4))) int v4i_t;

#define TSEQ   2048
#define DMODEL 512
#define NHEADS 8
#define DHEAD  64
#define BATCH  2

__device__ __forceinline__ int lane_id() { return (int)(threadIdx.x & 31u); }

// ---- WMMA fragment loaders (CDNA5 ISA 7.12.2 layouts, bf16 16x16x32) ----
// A (16x32, MxK): lane l: m=l&15, hi=l>>4; elems 0..7 -> K=hi*8+0..7, elems 8..15 -> K=16+hi*8+0..7
__device__ __forceinline__ v16bf load_a16(const bf16_t* p, int ld) {
  const int lane = lane_id();
  const int m = lane & 15, hi = lane >> 4;
  const bf16_t* row = p + m * ld + hi * 8;
  union { v16bf v; v8bf h[2]; } u;
  u.h[0] = *(const v8bf*)(row);
  u.h[1] = *(const v8bf*)(row + 16);
  return u.v;
}
// B (32x16, KxN) read from an N-major (transposed) buffer: lane l: n=l&15, hi=l>>4; elems 0..7 -> K=hi*8+0..7, elems 8..15 -> K=16+hi*8+0..7
__device__ __forceinline__ v16bf load_b16(const bf16_t* p, int ld) {
  const int lane = lane_id();
  const int n = lane & 15, hi = lane >> 4;
  const bf16_t* col = p + n * ld + hi * 8;
  union { v16bf v; v8bf h[2]; } u;
  u.h[0] = *(const v8bf*)(col);
  u.h[1] = *(const v8bf*)(col + 16);
  return u.v;
}

__device__ __forceinline__ v8f wmma_bf16(v16bf a, v16bf b, v8f c) {
  return __builtin_amdgcn_wmma_f32_16x16x32_bf16(false, a, false, b, (short)0, c, false, false);
}

__device__ __forceinline__ float rowmax16(float x) {
  x = fmaxf(x, __shfl_xor(x, 1, 32));
  x = fmaxf(x, __shfl_xor(x, 2, 32));
  x = fmaxf(x, __shfl_xor(x, 4, 32));
  x = fmaxf(x, __shfl_xor(x, 8, 32));
  return x;
}
__device__ __forceinline__ float rowsum16(float x) {
  x += __shfl_xor(x, 1, 32);
  x += __shfl_xor(x, 2, 32);
  x += __shfl_xor(x, 4, 32);
  x += __shfl_xor(x, 8, 32);
  return x;
}

// ---- Tensor Data Mover descriptor (ISA ch.8, D#) ----
__device__ __forceinline__ void set_field(unsigned int* d, int lo, int width, unsigned long long val) {
  const int w = lo >> 5, s = lo & 31;
  d[w] |= (unsigned int)(val << s);
  if (s + width > 32) d[w + 1] |= (unsigned int)(val >> (32 - s));
}

// 2D tile load: tile0 elems per row, tile1 rows (0 = 1-D), row stride in elements, 2-byte elems
__device__ __forceinline__ void tdm_load_2d(const void* gptr, const void* lptr,
                                            unsigned int tile0, unsigned int tile1,
                                            unsigned long long stride0) {
  unsigned int d0[4] = {0, 0, 0, 0};
  unsigned int d1[8] = {0, 0, 0, 0, 0, 0, 0, 0};
  const unsigned long long ga = (unsigned long long)(uintptr_t)gptr;
  const unsigned int lds = (unsigned int)(uintptr_t)lptr;   // flat->LDS: low 32 bits
  d0[0] = 1u;                                  // count = 1 valid descriptor
  d0[1] = lds;                                 // lds_addr [63:32]
  set_field(d0, 64, 57, ga);                   // global_addr [120:64]
  d0[3] |= (2u << 30);                         // type = 2 ("image")
  set_field(d1, 16, 2, 1);                     // data_size = 2 bytes
  set_field(d1, 48, 32, 0x7FFFFFFFull);        // tensor_dim0 (no OOB clip)
  set_field(d1, 80, 32, 0x7FFFFFFFull);        // tensor_dim1
  set_field(d1, 112, 16, tile0);               // tile_dim0
  set_field(d1, 128, 16, tile1);               // tile_dim1 (0 = unused)
  set_field(d1, 160, 48, stride0);             // tensor_dim0_stride
  v4u g0; v8i_t g1; v4i_t g2 = {0, 0, 0, 0}, g3 = {0, 0, 0, 0};
#pragma unroll
  for (int i = 0; i < 4; ++i) g0[i] = d0[i];
#pragma unroll
  for (int i = 0; i < 8; ++i) g1[i] = (int)d1[i];
#if __clang_major__ >= 23
  v8i_t g4 = {0, 0, 0, 0, 0, 0, 0, 0};
  __builtin_amdgcn_tensor_load_to_lds(g0, g1, g2, g3, g4, 0);
#else
  __builtin_amdgcn_tensor_load_to_lds(g0, g1, g2, g3, 0);
#endif
}

// ---------------- prep kernels ----------------
__global__ void cvt_bf16_kernel(const float* __restrict__ src, bf16_t* __restrict__ dst, int n) {
  int i = blockIdx.x * blockDim.x + threadIdx.x;
  if (i < n) dst[i] = (bf16_t)src[i];
}
__global__ void transpose_bf16_kernel(const float* __restrict__ src, bf16_t* __restrict__ dst,
                                      int rows, int cols) {
  int i = blockIdx.x * blockDim.x + threadIdx.x;
  if (i < rows * cols) {
    int r = i / cols, c = i - r * cols;
    dst[(size_t)c * rows + r] = (bf16_t)src[i];
  }
}
__global__ void relbias_kernel(const float* __restrict__ table, const float* __restrict__ w,
                               float* __restrict__ out) {
  int i = blockIdx.x * blockDim.x + threadIdx.x;
  if (i < 2 * TSEQ - 1) {
    float s = 0.f;
#pragma unroll 16
    for (int d = 0; d < DHEAD; ++d) s += table[i * DHEAD + d] * w[d];
    out[i] = s;
  }
}

// ---------------- QKV GEMM: (4096x512) @ (512x1536) ----------------
__global__ __launch_bounds__(256) void qkv_kernel(const bf16_t* __restrict__ xb,
                                                  const bf16_t* __restrict__ Wt,
                                                  const float* __restrict__ bias,
                                                  bf16_t* __restrict__ Q,
                                                  bf16_t* __restrict__ K,
                                                  bf16_t* __restrict__ Vt) {
  const int wave = threadIdx.x >> 5;
  const int tile = blockIdx.x * 8 + wave;          // 6144 tiles
  const int row0 = (tile / 24) * 16;               // M = 4096
  const int col0 = (tile % 24) * 64;               // N = 1536
  v8f acc[4] = {};
  for (int k = 0; k < DMODEL; k += 32) {
    v16bf a = load_a16(xb + (size_t)row0 * DMODEL + k, DMODEL);
#pragma unroll
    for (int j = 0; j < 4; ++j) {
      v16bf b = load_b16(Wt + (size_t)(col0 + 16 * j) * DMODEL + k, DMODEL);
      acc[j] = wmma_bf16(a, b, acc[j]);
    }
  }
  const int lane = lane_id();
  const int hi = lane >> 4, n = lane & 15;
#pragma unroll
  for (int j = 0; j < 4; ++j) {
    const int c = col0 + 16 * j + n;
    const float bv = bias[c];
    const int sel = c >> 9;                 // 0=q 1=k 2=v
    const int d = c & 511;
    const int h = d >> 6, dh = d & 63;
#pragma unroll
    for (int r = 0; r < 8; ++r) {
      const int grow = row0 + r + 8 * hi;   // b*T + t
      const int b = grow >> 11, t = grow & (TSEQ - 1);
      const float val = acc[j][r] + bv;
      const size_t bh = (size_t)(b * NHEADS + h);
      if (sel == 0)      Q[(bh * TSEQ + t) * DHEAD + dh] = (bf16_t)val;
      else if (sel == 1) K[(bh * TSEQ + t) * DHEAD + dh] = (bf16_t)val;
      else               Vt[(bh * DHEAD + dh) * TSEQ + t] = (bf16_t)val;
    }
  }
}

// ---------------- Flash attention with TDM-staged K/V tiles ----------------
// LDS layout (single array so offsets are known for the D#):
//   [0,      4096) elems : K tiles, double buffered (2 x 32x64)
//   [4096,   8192) elems : V tiles, double buffered (2 x 64x32)
//   [8192,  12288) elems : per-wave P tiles (8 x 16x32)
__global__ __launch_bounds__(256) void attn_kernel(const bf16_t* __restrict__ Q,
                                                   const bf16_t* __restrict__ K,
                                                   const bf16_t* __restrict__ Vt,
                                                   const float* __restrict__ relbias,
                                                   const unsigned char* __restrict__ mask,
                                                   bf16_t* __restrict__ Obf) {
  __shared__ __attribute__((aligned(128))) bf16_t smem[12288];
  const int wave = threadIdx.x >> 5;
  const int lane = lane_id();
  const int hi = lane >> 4, n = lane & 15;
  const int bh = blockIdx.y;                 // B*H = 16
  const int b = bh >> 3, h = bh & 7;
  const int q0 = blockIdx.x * 128 + wave * 16;
  const bf16_t* Qb = Q  + (size_t)bh * TSEQ * DHEAD;
  const bf16_t* Kb = K  + (size_t)bh * TSEQ * DHEAD;
  const bf16_t* Vb = Vt + (size_t)bh * DHEAD * TSEQ;
  const float scale = 0.125f;                // Dh^-0.5

  const v16bf aq0 = load_a16(Qb + (size_t)q0 * DHEAD + 0,  DHEAD);
  const v16bf aq1 = load_a16(Qb + (size_t)q0 * DHEAD + 32, DHEAD);

  v8f o[4] = {};
  float mrow[8], lrow[8];
#pragma unroll
  for (int r = 0; r < 8; ++r) { mrow[r] = -1e30f; lrow[r] = 0.f; }

  // preload key-tile 0 into buffer 0 (K tile is contiguous 2048 elems; V tile is 64 rows x 32)
  if (wave == 0) {
    tdm_load_2d(Kb, smem, 2048u, 0u, 2048ull);
    tdm_load_2d(Vb, smem + 4096, 32u, 64u, (unsigned long long)TSEQ);
  }

  for (int j0 = 0; j0 < TSEQ; j0 += 32) {
    const int buf = (j0 >> 5) & 1;
    if (wave == 0) {
      if (j0 + 32 < TSEQ) {
        const int nb = buf ^ 1;
        tdm_load_2d(Kb + (size_t)(j0 + 32) * DHEAD, smem + nb * 2048, 2048u, 0u, 2048ull);
        tdm_load_2d(Vb + (j0 + 32), smem + 4096 + nb * 2048, 32u, 64u, (unsigned long long)TSEQ);
        __builtin_amdgcn_s_wait_tensorcnt(2);   // current tile's 2 loads done
      } else {
        __builtin_amdgcn_s_wait_tensorcnt(0);
      }
    }
    __syncthreads();

    const bf16_t* kb = smem + buf * 2048;          // 32 x 64, row-major
    const bf16_t* vb = smem + 4096 + buf * 2048;   // 64 x 32 (d-major)

    // S = Q K^T for 32 keys (two 16x16 C tiles)
    v8f s0 = {}, s1 = {};
    s0 = wmma_bf16(aq0, load_b16(kb + 0,            DHEAD), s0);
    s0 = wmma_bf16(aq1, load_b16(kb + 32,           DHEAD), s0);
    s1 = wmma_bf16(aq0, load_b16(kb + 16 * DHEAD,      DHEAD), s1);
    s1 = wmma_bf16(aq1, load_b16(kb + 16 * DHEAD + 32, DHEAD), s1);

    float p0[8], p1[8], corr[8];
#pragma unroll
    for (int r = 0; r < 8; ++r) {
      const int qi  = q0 + r + 8 * hi;
      const int k0i = j0 + n, k1i = j0 + 16 + n;
      float v0 = s0[r] * scale + relbias[qi - k0i + (TSEQ - 1)];
      float v1 = s1[r] * scale + relbias[qi - k1i + (TSEQ - 1)];
      if (!mask[b * TSEQ + k0i]) v0 = -1e30f;
      if (!mask[b * TSEQ + k1i]) v1 = -1e30f;
      const float mx   = rowmax16(fmaxf(v0, v1));
      const float mnew = fmaxf(mrow[r], mx);
      const float c    = __expf(mrow[r] - mnew);
      p0[r] = __expf(v0 - mnew);
      p1[r] = __expf(v1 - mnew);
      lrow[r] = lrow[r] * c + rowsum16(p0[r] + p1[r]);
      mrow[r] = mnew;
      corr[r] = c;
    }

    // re-fragment P (C-layout -> A-layout) via per-wave LDS tile
    bf16_t* pt = smem + 8192 + wave * 512;
#pragma unroll
    for (int r = 0; r < 8; ++r) {
      const int m = r + 8 * hi;
      pt[m * 32 + n]      = (bf16_t)p0[r];
      pt[m * 32 + 16 + n] = (bf16_t)p1[r];
    }
    asm volatile("s_wait_dscnt 0" ::: "memory");  // intra-wave LDS RAW (DS in-order)
    const v16bf pa = load_a16(pt, 32);
    asm volatile("" ::: "memory");

    // O = diag(corr)*O + P V
#pragma unroll
    for (int j = 0; j < 4; ++j) {
#pragma unroll
      for (int r = 0; r < 8; ++r) o[j][r] *= corr[r];
      v16bf bv = load_b16(vb + (j * 16) * 32, 32);
      o[j] = wmma_bf16(pa, bv, o[j]);
    }
    __syncthreads();   // protect LDS buffers before next TDM overwrite
  }

#pragma unroll
  for (int j = 0; j < 4; ++j)
#pragma unroll
    for (int r = 0; r < 8; ++r) {
      const int t = q0 + r + 8 * hi;
      const int d = h * DHEAD + j * 16 + n;
      Obf[((size_t)(b * TSEQ + t)) * DMODEL + d] = (bf16_t)(o[j][r] / lrow[r]);
    }
}

// ---------------- Output projection: (4096x512) @ (512x512) + b, fp32 out ----------------
__global__ __launch_bounds__(256) void proj_kernel(const bf16_t* __restrict__ Ob,
                                                   const bf16_t* __restrict__ Wt,
                                                   const float* __restrict__ bias,
                                                   float* __restrict__ out) {
  const int wave = threadIdx.x >> 5;
  const int tile = blockIdx.x * 8 + wave;          // 2048 tiles
  const int row0 = (tile / 8) * 16;
  const int col0 = (tile % 8) * 64;
  v8f acc[4] = {};
  for (int k = 0; k < DMODEL; k += 32) {
    v16bf a = load_a16(Ob + (size_t)row0 * DMODEL + k, DMODEL);
#pragma unroll
    for (int j = 0; j < 4; ++j) {
      v16bf b = load_b16(Wt + (size_t)(col0 + 16 * j) * DMODEL + k, DMODEL);
      acc[j] = wmma_bf16(a, b, acc[j]);
    }
  }
  const int lane = lane_id();
  const int hi = lane >> 4, n = lane & 15;
#pragma unroll
  for (int j = 0; j < 4; ++j) {
    const int c = col0 + 16 * j + n;
    const float bv = bias[c];
#pragma unroll
    for (int r = 0; r < 8; ++r)
      out[(size_t)(row0 + r + 8 * hi) * DMODEL + c] = acc[j][r] + bv;
  }
}

extern "C" void kernel_launch(void* const* d_in, const int* in_sizes, int n_in,
                              void* d_out, int out_size, void* d_ws, size_t ws_size,
                              hipStream_t stream) {
  const float* x      = (const float*)d_in[0];
  const unsigned char* mask = (const unsigned char*)d_in[1];
  const float* W_qkv  = (const float*)d_in[2];
  const float* b_qkv  = (const float*)d_in[3];
  const float* W_proj = (const float*)d_in[4];
  const float* b_proj = (const float*)d_in[5];
  const float* rpe_t  = (const float*)d_in[6];
  const float* rpe_w  = (const float*)d_in[7];
  float* out = (float*)d_out;

  char* ws = (char*)d_ws;
  size_t off = 0;
  auto take = [&](size_t bytes) { char* p = ws + off; off += (bytes + 255) & ~(size_t)255; return p; };
  bf16_t* xb     = (bf16_t*)take((size_t)BATCH * TSEQ * DMODEL * 2);
  bf16_t* Wqkvt  = (bf16_t*)take((size_t)3 * DMODEL * DMODEL * 2);
  bf16_t* Wprjt  = (bf16_t*)take((size_t)DMODEL * DMODEL * 2);
  bf16_t* Qw     = (bf16_t*)take((size_t)BATCH * NHEADS * TSEQ * DHEAD * 2);
  bf16_t* Kw     = (bf16_t*)take((size_t)BATCH * NHEADS * TSEQ * DHEAD * 2);
  bf16_t* Vtw    = (bf16_t*)take((size_t)BATCH * NHEADS * TSEQ * DHEAD * 2);
  bf16_t* Obf    = (bf16_t*)take((size_t)BATCH * TSEQ * DMODEL * 2);
  float*  relb   = (float*)take((size_t)(2 * TSEQ - 1) * 4);

  const int nx = BATCH * TSEQ * DMODEL;
  cvt_bf16_kernel<<<(nx + 255) / 256, 256, 0, stream>>>(x, xb, nx);
  transpose_bf16_kernel<<<(DMODEL * 3 * DMODEL + 255) / 256, 256, 0, stream>>>(W_qkv, Wqkvt, DMODEL, 3 * DMODEL);
  transpose_bf16_kernel<<<(DMODEL * DMODEL + 255) / 256, 256, 0, stream>>>(W_proj, Wprjt, DMODEL, DMODEL);
  relbias_kernel<<<(2 * TSEQ - 1 + 255) / 256, 256, 0, stream>>>(rpe_t, rpe_w, relb);

  qkv_kernel<<<768, 256, 0, stream>>>(xb, Wqkvt, b_qkv, Qw, Kw, Vtw);
  attn_kernel<<<dim3(TSEQ / 128, BATCH * NHEADS), 256, 0, stream>>>(Qw, Kw, Vtw, relb, mask, Obf);
  proj_kernel<<<256, 256, 0, stream>>>(Obf, Wprjt, b_proj, out);
}
